// GAT_40278203301987
// MI455X (gfx1250) — matrix operand order repaired and checked by reading, not compiled, not run on hardware
//
#include <hip/hip_runtime.h>
#include <hip/hip_bf16.h>

// ---------------- problem constants (from reference) ----------------
#define N_COMBIN 50000
#define N_DEV    100000
#define CONT     62
#define VOCAB    1000
#define EMB      16
#define BSZ      8192
#define KNB      64           // neighbors
#define NH       4            // heads
#define OUTD     16           // per-head out
#define NCLS     1
#define FEAT     94           // CONT + 2*EMB
#define FEATP    96           // padded to 3 * 32 for WMMA k-steps
#define HO       64           // NH * OUTD
#define FUSD     204          // OUT + 2*FEAT
#define ALPHA    0.2f
#define HNS      17           // h_n LDS row stride (pad vs 64-bank conflicts)

typedef __attribute__((ext_vector_type(16))) _Float16 v16h;
typedef __attribute__((ext_vector_type(8)))  _Float16 v8h;
typedef __attribute__((ext_vector_type(8)))  float    v8f;

// ---------------- workspace layout (bytes) ----------------
#define OFF_DEVF16  ((size_t)0)                       // N_DEV*96 f16   = 19,200,000
#define OFF_COMBF16 ((size_t)19200000)                // B*96 f16       =  1,572,864
#define OFF_COMBF32 ((size_t)20772864)                // B*96 f32       =  3,145,728
#define OFF_DEVBF32 ((size_t)23918592)                // B*96 f32       =  3,145,728
#define OFF_WCT     ((size_t)27064320)                // 64*96 f16      =     12,288
#define OFF_WDT     ((size_t)27076608)                // 64*96 f16      =     12,288
#define OFF_HC      ((size_t)27088896)                // B*64 f32       =  2,097,152

// ---------------- wave32 reductions ----------------
__device__ __forceinline__ float wave_sum(float v) {
  #pragma unroll
  for (int m = 16; m >= 1; m >>= 1) v += __shfl_xor(v, m, 32);
  return v;
}
__device__ __forceinline__ float wave_max(float v) {
  #pragma unroll
  for (int m = 16; m >= 1; m >>= 1) v = fmaxf(v, __shfl_xor(v, m, 32));
  return v;
}

// A-fragment (16x32 f16): lane (half h) holds row m=lane%16,
// K = kb..kb+15 in v[0..15], kb = 32*kk + 16*half.
__device__ __forceinline__ v16h load_a_frag(const _Float16* __restrict__ rowp, int kb) {
  v8h lo = *(const v8h*)(rowp + kb);
  v8h hi = *(const v8h*)(rowp + kb + 8);
  v16h a;
  #pragma unroll
  for (int i = 0; i < 8; ++i) { a[i] = lo[i]; a[i + 8] = hi[i]; }
  return a;
}

// ---------------- 1. weight prep: Wc/Wd (h,f,o) f32 -> WT[n=h*16+o][f] f16, f padded to 96
__global__ void __launch_bounds__(256) prep_weights_kernel(
    const float* __restrict__ Wc, const float* __restrict__ Wd,
    _Float16* __restrict__ WcT, _Float16* __restrict__ WdT) {
  int idx = blockIdx.x * 256 + threadIdx.x;
  if (idx >= HO * FEATP) return;
  int n = idx / FEATP, f = idx % FEATP;
  int h = n >> 4, o = n & 15;
  float vc = 0.f, vd = 0.f;
  if (f < FEAT) {
    size_t s = ((size_t)h * FEAT + f) * OUTD + o;
    vc = Wc[s]; vd = Wd[s];
  }
  WcT[idx] = (_Float16)vc;
  WdT[idx] = (_Float16)vd;
}

// ---------------- 2. dev_feats table: (N_DEV,96) f16, one wave per row
__global__ void __launch_bounds__(256) build_dev_feats_kernel(
    const float* __restrict__ dc, const int* __restrict__ dcat,
    const float* __restrict__ de0, const float* __restrict__ de1,
    _Float16* __restrict__ outF16) {
  int wave = threadIdx.x >> 5, lane = threadIdx.x & 31;
  int r = blockIdx.x * 8 + wave;
  if (r >= N_DEV) return;
  int c0 = dcat[r * 2 + 0], c1 = dcat[r * 2 + 1];
  _Float16* row = outF16 + (size_t)r * FEATP;
  for (int c = lane; c < FEATP; c += 32) {
    float v;
    if (c < CONT)            v = dc[(size_t)r * CONT + c];
    else if (c < CONT + EMB) v = de0[c0 * EMB + (c - CONT)];
    else if (c < FEAT)       v = de1[c1 * EMB + (c - CONT - EMB)];
    else                     v = 0.f;
    row[c] = (_Float16)v;
  }
}

// ---------------- 3. batch gathers: comb f16+f32, dev_batch f32; one wave per row
__global__ void __launch_bounds__(256) build_batch_kernel(
    const float* __restrict__ cc, const int* __restrict__ ccat,
    const float* __restrict__ ce0, const float* __restrict__ ce1,
    const float* __restrict__ dc, const int* __restrict__ dcat,
    const float* __restrict__ de0, const float* __restrict__ de1,
    const int* __restrict__ cidx, const int* __restrict__ didx,
    _Float16* __restrict__ combF16, float* __restrict__ combF32,
    float* __restrict__ devbF32) {
  int wave = threadIdx.x >> 5, lane = threadIdx.x & 31;
  int b = blockIdx.x * 8 + wave;
  if (b >= BSZ) return;
  int ci = cidx[b], di = didx[b];
  int cc0 = ccat[ci * 2], cc1 = ccat[ci * 2 + 1];
  int dc0 = dcat[di * 2], dc1 = dcat[di * 2 + 1];
  for (int c = lane; c < FEATP; c += 32) {
    float cv, dv;
    if (c < CONT)            { cv = cc[(size_t)ci * CONT + c];        dv = dc[(size_t)di * CONT + c]; }
    else if (c < CONT + EMB) { cv = ce0[cc0 * EMB + (c - CONT)];      dv = de0[dc0 * EMB + (c - CONT)]; }
    else if (c < FEAT)       { cv = ce1[cc1 * EMB + (c - CONT - EMB)]; dv = de1[dc1 * EMB + (c - CONT - EMB)]; }
    else                     { cv = 0.f; dv = 0.f; }
    combF16[(size_t)b * FEATP + c] = (_Float16)cv;
    combF32[(size_t)b * FEATP + c] = cv;
    devbF32[(size_t)b * FEATP + c] = dv;
  }
}

// ---------------- 4. h_c = comb(8192x96) @ WcT(96x64) + bc via WMMA
__global__ void __launch_bounds__(128) hc_gemm_kernel(
    const _Float16* __restrict__ combF16, const _Float16* __restrict__ WcT,
    const float* __restrict__ bc, float* __restrict__ h_c) {
  __shared__ _Float16 sW[HO * FEATP];
  const int tid = threadIdx.x;
  for (int i = tid * 8; i < HO * FEATP; i += 128 * 8)
    *(v8h*)(sW + i) = *(const v8h*)(WcT + i);
  __syncthreads();

  const int wave = tid >> 5, lane = tid & 31;
  const int lr = lane & 15, half = lane >> 4;
  const int m0 = (blockIdx.x * 4 + wave) * 16;
  const _Float16* arow = combF16 + (size_t)(m0 + lr) * FEATP;

  v16h a[3];
  #pragma unroll
  for (int kk = 0; kk < 3; ++kk) a[kk] = load_a_frag(arow, kk * 32 + half * 16);

  #pragma unroll
  for (int nt = 0; nt < 4; ++nt) {
    const int n = nt * 16 + lr;
    const float bias = bc[n];
    v8f acc;
    #pragma unroll
    for (int r = 0; r < 8; ++r) acc[r] = bias;
    #pragma unroll
    for (int kk = 0; kk < 3; ++kk) {
      v16h bt = *(const v16h*)(sW + n * FEATP + kk * 32 + half * 16);
      acc = __builtin_amdgcn_wmma_f32_16x16x32_f16(false, a[kk], false, bt,
                                                   (short)0, acc, false, false);
    }
    #pragma unroll
    for (int r = 0; r < 8; ++r)
      h_c[(size_t)(m0 + r + half * 8) * HO + n] = acc[r];
  }
}

// ---------------- 5. fused GAT: one wave = one batch row ----------------
__global__ void __launch_bounds__(128) gat_fused_kernel(
    const _Float16* __restrict__ devF16, const _Float16* __restrict__ WdT,
    const int* __restrict__ neighbor_idx, const float* __restrict__ h_c,
    const float* __restrict__ bd, const float* __restrict__ aW,
    const float* __restrict__ ab, const float* __restrict__ W1,
    const float* __restrict__ b1, const float* __restrict__ combF32,
    const float* __restrict__ devbF32, const float* __restrict__ W2,
    const float* __restrict__ b2, const float* __restrict__ W3,
    const float* __restrict__ b3, const float* __restrict__ W4,
    const float* __restrict__ b4, float* __restrict__ out) {
  __shared__ _Float16 sW[HO * FEATP];               // 12 KB, shared by block
  __shared__ float s_hn[4][KNB][HNS];               // per-wave h_n tile
  __shared__ float s_attn[4][KNB];
  __shared__ float s_ho[4][HO];
  __shared__ float s_af[4][OUTD];
  __shared__ float s_fus[4][FUSD + 4];
  __shared__ float s_x2[4][64];
  __shared__ float s_ec[4][NH];

  const int tid = threadIdx.x;
  for (int i = tid * 8; i < HO * FEATP; i += 128 * 8)
    *(v8h*)(sW + i) = *(const v8h*)(WdT + i);
  __syncthreads();

  const int w = tid >> 5, lane = tid & 31;
  const int lr = lane & 15, half = lane >> 4;
  const int b = blockIdx.x * 4 + w;

  // e_c[h] = <h_c[b,h,:], a1[h]>  (a1 = aW[:, :16])
  #pragma unroll
  for (int h = 0; h < NH; ++h) {
    float p = (half == 0) ? h_c[(size_t)b * HO + h * OUTD + lr] * aW[h * 2 * OUTD + lr] : 0.f;
    float s = wave_sum(p);
    if (lane == 0) s_ec[w][h] = s;
  }

  // A fragments: 64 neighbor rows x 96 f16, kept in VGPRs for all heads
  v16h a[4][3];
  #pragma unroll
  for (int mt = 0; mt < 4; ++mt) {
    const int m = mt * 16 + lr;
    const int nb = neighbor_idx[b * KNB + m];
    const _Float16* rowp = devF16 + (size_t)nb * FEATP;
    #pragma unroll
    for (int kk = 0; kk < 3; ++kk)
      a[mt][kk] = load_a_frag(rowp, kk * 32 + half * 16);
  }

  for (int h = 0; h < NH; ++h) {
    // h_n = neibr @ Wd[h] + bd[h]  (bias folded into C init)
    const float bias = bd[h * OUTD + lr];
    v8f acc[4];
    #pragma unroll
    for (int mt = 0; mt < 4; ++mt) {
      #pragma unroll
      for (int r = 0; r < 8; ++r) acc[mt][r] = bias;
    }
    #pragma unroll
    for (int kk = 0; kk < 3; ++kk) {
      v16h bt = *(const v16h*)(sW + (h * OUTD + lr) * FEATP + kk * 32 + half * 16);
      #pragma unroll
      for (int mt = 0; mt < 4; ++mt)
        acc[mt] = __builtin_amdgcn_wmma_f32_16x16x32_f16(false, a[mt][kk], false, bt,
                                                         (short)0, acc[mt], false, false);
    }
    // spill h_n (64 x 16) to padded LDS
    #pragma unroll
    for (int mt = 0; mt < 4; ++mt) {
      #pragma unroll
      for (int r = 0; r < 8; ++r)
        s_hn[w][mt * 16 + r + half * 8][lr] = acc[mt][r];
    }

    // e_k = leaky(e_c + <h_n[k], a2[h]> + ab[h]); softmax over k (2 k per lane)
    const float e0 = s_ec[w][h] + ab[h];
    float ea = e0, eb = e0;
    #pragma unroll
    for (int o = 0; o < OUTD; ++o) {
      const float wgt = aW[h * 2 * OUTD + OUTD + o];
      ea += s_hn[w][lane][o] * wgt;
      eb += s_hn[w][lane + 32][o] * wgt;
    }
    ea = (ea >= 0.f) ? ea : ALPHA * ea;
    eb = (eb >= 0.f) ? eb : ALPHA * eb;
    const float mx = wave_max(fmaxf(ea, eb));
    const float xa = __expf(ea - mx), xb = __expf(eb - mx);
    const float inv = 1.f / wave_sum(xa + xb);
    s_attn[w][lane] = xa * inv;
    s_attn[w][lane + 32] = xb * inv;

    // head_out[o] = elu(sum_k attn[k] * h_n[k][o]); halves split the k range
    float ho = 0.f;
    for (int k = half * 32; k < half * 32 + 32; ++k)
      ho += s_attn[w][k] * s_hn[w][k][lr];
    ho += __shfl_xor(ho, 16, 32);
    ho = (ho > 0.f) ? ho : (__expf(ho) - 1.f);
    if (half == 0) s_ho[w][h * OUTD + lr] = ho;
  }

  // attn_feats = head_out(64) @ W1(64x16) + b1
  float af = 0.f;
  for (int j = half * 32; j < half * 32 + 32; ++j)
    af += s_ho[w][j] * W1[j * OUTD + lr];
  af += __shfl_xor(af, 16, 32);
  af += b1[lr];
  if (half == 0) s_af[w][lr] = af;

  // fusion vector [comb(94) | dev(94) | attn_feats(16)]
  const float* crow = combF32 + (size_t)b * FEATP;
  const float* drow = devbF32 + (size_t)b * FEATP;
  for (int i = lane; i < FEAT; i += 32) {
    s_fus[w][i] = crow[i];
    s_fus[w][FEAT + i] = drow[i];
  }
  if (lane < OUTD) s_fus[w][2 * FEAT + lane] = s_af[w][lane];

  // x2 = relu(fus @ W2(204x64) + b2): 2 outputs per lane
  float a2a = b2[lane], a2b = b2[lane + 32];
  for (int i = 0; i < FUSD; ++i) {
    const float fi = s_fus[w][i];
    a2a += fi * W2[i * 64 + lane];
    a2b += fi * W2[i * 64 + lane + 32];
  }
  s_x2[w][lane] = fmaxf(a2a, 0.f);
  s_x2[w][lane + 32] = fmaxf(a2b, 0.f);

  // x3 = relu(x2 @ W3(64x32) + b3): 1 output per lane
  float a3 = b3[lane];
  for (int i = 0; i < 64; ++i) a3 += s_x2[w][i] * W3[i * 32 + lane];
  a3 = fmaxf(a3, 0.f);

  // out = sigmoid(x3 @ W4(32x1) + b4)
  const float s = wave_sum(a3 * W4[lane]);
  if (lane == 0) out[b] = 1.f / (1.f + __expf(-(s + b4[0])));
}

// ---------------- host launcher ----------------
extern "C" void kernel_launch(void* const* d_in, const int* in_sizes, int n_in,
                              void* d_out, int out_size, void* d_ws, size_t ws_size,
                              hipStream_t stream) {
  (void)in_sizes; (void)n_in; (void)out_size; (void)ws_size;
  const float* combin_cont  = (const float*)d_in[0];
  const int*   combin_cat   = (const int*)d_in[1];
  const float* device_cont  = (const float*)d_in[2];
  const int*   device_cat   = (const int*)d_in[3];
  const int*   combin_idx   = (const int*)d_in[4];
  const int*   device_idx   = (const int*)d_in[5];
  const int*   neighbor_idx = (const int*)d_in[6];
  const float* ce0 = (const float*)d_in[7];
  const float* ce1 = (const float*)d_in[8];
  const float* de0 = (const float*)d_in[9];
  const float* de1 = (const float*)d_in[10];
  const float* Wc  = (const float*)d_in[11];
  const float* bc  = (const float*)d_in[12];
  const float* Wd  = (const float*)d_in[13];
  const float* bd  = (const float*)d_in[14];
  const float* aW  = (const float*)d_in[15];
  const float* ab  = (const float*)d_in[16];
  const float* W1  = (const float*)d_in[17];
  const float* b1  = (const float*)d_in[18];
  const float* W2  = (const float*)d_in[19];
  const float* b2  = (const float*)d_in[20];
  const float* W3  = (const float*)d_in[21];
  const float* b3  = (const float*)d_in[22];
  const float* W4  = (const float*)d_in[23];
  const float* b4  = (const float*)d_in[24];

  char* ws = (char*)d_ws;
  _Float16* devF16  = (_Float16*)(ws + OFF_DEVF16);
  _Float16* combF16 = (_Float16*)(ws + OFF_COMBF16);
  float*    combF32 = (float*)(ws + OFF_COMBF32);
  float*    devbF32 = (float*)(ws + OFF_DEVBF32);
  _Float16* WcT     = (_Float16*)(ws + OFF_WCT);
  _Float16* WdT     = (_Float16*)(ws + OFF_WDT);
  float*    h_c     = (float*)(ws + OFF_HC);
  float*    out     = (float*)d_out;

  prep_weights_kernel<<<(HO * FEATP + 255) / 256, 256, 0, stream>>>(Wc, Wd, WcT, WdT);
  build_dev_feats_kernel<<<(N_DEV + 7) / 8, 256, 0, stream>>>(
      device_cont, device_cat, de0, de1, devF16);
  build_batch_kernel<<<(BSZ + 7) / 8, 256, 0, stream>>>(
      combin_cont, combin_cat, ce0, ce1, device_cont, device_cat, de0, de1,
      combin_idx, device_idx, combF16, combF32, devbF32);
  hc_gemm_kernel<<<BSZ / 64, 128, 0, stream>>>(combF16, WcT, bc, h_c);
  gat_fused_kernel<<<BSZ / 4, 128, 0, stream>>>(
      devF16, WdT, neighbor_idx, h_c, bd, aW, ab, W1, b1, combF32, devbF32,
      W2, b2, W3, b3, W4, b4, out);
}
